// OuterProductMean_23476291239989
// MI455X (gfx1250) — matrix operand order repaired and checked by reading, not compiled, not run on hardware
//
#include <hip/hip_runtime.h>

// ---------------- problem constants (from reference) ----------------
#define B_  1
#define R_  256
#define S_  128
#define C_  256
#define CO_ 32
#define OUT_ 128
#define LN_EPS 1e-5f
#define EPS_   1e-3f

// padded LDS row strides (halves): 272B / 2064B rows -> banks 4*lane mod 64
#define PANW 136            // panel row stride (128 data + 8 pad)
#define OPW  1032           // opbuf row stride (1024 data + 8 pad)
#define PAN_ROWS 256        // 128 left rows + 128 right rows
#define OP_OFF   (PAN_ROWS * PANW)           // 34816 halves
#define SCL_OFF  (OP_OFF + 16 * OPW)         // 51328 halves (float-aligned)
#define SMEM_BYTES (SCL_OFF * 2 + 16 * 4)    // 102720 bytes

typedef __attribute__((ext_vector_type(16))) _Float16 v16h;
typedef __attribute__((ext_vector_type(8)))  _Float16 v8h;
typedef __attribute__((ext_vector_type(8)))  float    v8f;
typedef __attribute__((ext_vector_type(4)))  float    float4v;

static __device__ __forceinline__ v8h ldg_h8(const _Float16* p) {
    return *(const v8h*)p;
}
static __device__ __forceinline__ v16h mk16(v8h a, v8h b) {
    v16h r;
    #pragma unroll
    for (int e = 0; e < 8; ++e) { r[e] = a[e]; r[8 + e] = b[e]; }
    return r;
}

// =====================================================================
// Kernel 0: convert fp32 weights -> f16 panels in workspace.
//  wlr[64][256]  : rows 0..31 = w_left, rows 32..63 = w_right (B-layout)
//  woT[128][1024]: w_o rows as-is (B-layout for GEMM2)
// =====================================================================
__global__ void k_prep(const float* __restrict__ w_left,
                       const float* __restrict__ w_right,
                       const float* __restrict__ w_o,
                       _Float16* __restrict__ wlr,
                       _Float16* __restrict__ woT) {
    const int NW = 64 * C_;          // 16384
    const int NO = OUT_ * CO_ * CO_; // 131072
    for (int idx = blockIdx.x * blockDim.x + threadIdx.x;
         idx < NW + NO; idx += gridDim.x * blockDim.x) {
        if (idx < NW) {
            int row = idx >> 8, col = idx & (C_ - 1);
            float v = (row < CO_) ? w_left[row * C_ + col]
                                  : w_right[(row - CO_) * C_ + col];
            wlr[idx] = (_Float16)v;
        } else {
            woT[idx - NW] = (_Float16)w_o[idx - NW];
        }
    }
}

// =====================================================================
// Kernel 1: fused LayerNorm + left/right projection (+bias, +mask).
// A-tile rows = 16 consecutive s for ONE i  => C-layout lanes hold 8
// consecutive s-values => packed 16B stores into the transposed panels:
//   leftT [i][c][s] = mask[s,i] * (LN(act[i,s,:]) . w_left[c,:]  + b_left[c])
//   rightT[i][d][s] = mask[s,i] * (LN(act[i,s,:]) . w_right[d,:] + b_right[d])
// grid = (S/16, R), block = 128 (4 waves; wave w = N-tile w of 64 wide).
// =====================================================================
__global__ void __launch_bounds__(128)
k_ln_proj(const float* __restrict__ act,   // [R][S][C]
          const float* __restrict__ mask,  // [S][R]
          const float* __restrict__ gamma,
          const float* __restrict__ beta,
          const _Float16* __restrict__ wlr, // [64][256]
          const float* __restrict__ b_left,
          const float* __restrict__ b_right,
          _Float16* __restrict__ leftT,     // [R][CO][S]
          _Float16* __restrict__ rightT) {  // [R][CO][S]
    const int s0 = blockIdx.x * 16;
    const int i  = blockIdx.y;
    const int tid = threadIdx.x;

    __shared__ _Float16 xn[16][C_ + 8];   // 528B rows: conflict-free, 16B-aligned
    __shared__ float redS[16][8], redQ[16][8];

    // ---- LayerNorm: 8 threads per row (row = s0 + r), 32 channels each ----
    const int r  = tid >> 3;
    const int q  = tid & 7;
    const int c0 = q * 32;
    const float* xrow = act + ((size_t)i * S_ + s0 + r) * C_;

    float4v xv[8];
    float sum = 0.f, sq = 0.f;
    #pragma unroll
    for (int t = 0; t < 8; ++t) {
        xv[t] = *(const float4v*)(xrow + c0 + t * 4);
        #pragma unroll
        for (int e = 0; e < 4; ++e) { float x = xv[t][e]; sum += x; sq += x * x; }
    }
    redS[r][q] = sum; redQ[r][q] = sq;
    __syncthreads();
    float ts = 0.f, tq = 0.f;
    #pragma unroll
    for (int t = 0; t < 8; ++t) { ts += redS[r][t]; tq += redQ[r][t]; }
    const float mu   = ts * (1.f / C_);
    const float var  = tq * (1.f / C_) - mu * mu;
    const float rstd = rsqrtf(var + LN_EPS);
    #pragma unroll
    for (int t = 0; t < 8; ++t)
        #pragma unroll
        for (int e = 0; e < 4; ++e) {
            int c = c0 + t * 4 + e;
            xn[r][c] = (_Float16)((xv[t][e] - mu) * rstd * gamma[c] + beta[c]);
        }
    __syncthreads();

    // ---- projection: one wave per 16-wide N-tile of the 64 channels ----
    const int wv   = tid >> 5;
    const int lane = tid & 31;
    const int lo16 = lane & 15;
    const int hi   = lane >> 4;
    const int ng   = wv * 16 + lo16;          // combined channel 0..63

    v8f acc = {};
    #pragma unroll
    for (int kc = 0; kc < 8; ++kc) {          // K = 256 in steps of 32
        const _Float16* pa = &xn[lo16][kc * 32 + hi * 8];
        v16h a = mk16(*(const v8h*)pa, *(const v8h*)(pa + 16));
        const _Float16* pb = wlr + ng * C_ + kc * 32 + hi * 8;
        v16h b = mk16(ldg_h8(pb), ldg_h8(pb + 16));
        acc = __builtin_amdgcn_wmma_f32_16x16x32_f16(false, a, false, b,
                                                     (short)0, acc, false, false);
    }

    const float bias = (wv < 2) ? b_left[ng] : b_right[ng - CO_];
    _Float16* dst = (wv < 2) ? leftT : rightT;
    const int co = (wv < 2) ? ng : (ng - CO_);
    // lane holds s = s0 + 8*hi + v  (v = 0..7): one packed 16B store
    v8h pk;
    #pragma unroll
    for (int v = 0; v < 8; ++v) {
        int s = s0 + 8 * hi + v;
        float mk = mask[(size_t)s * R_ + i];
        pk[v] = (_Float16)((acc[v] + bias) * mk);
    }
    *(v8h*)(dst + ((size_t)i * CO_ + co) * S_ + s0 + 8 * hi) = pk;
}

// =====================================================================
// Kernel 2: fused outer-product + output projection.
// grid = (R/4, R/4), block = 128 (4 waves). Tile = 4 i x 4 j = 16 pairs.
// Stage 64KB of left/right panels into LDS once (coalesced), then:
// GEMM1: op_p[c,d] = sum_s leftT[i][c][s]*rightT[j][d][s]  (wave w: i=i0+w)
// -> f16 LDS A-panel [16 pairs][1024] (padded rows)
// GEMM2: out[p][o] = sum_k op[p][k]*woT[o][k], + b_o, * 1/(eps+norm)
// Dynamic LDS: 256x136 panels + 16x1032 opbuf + 16 floats = ~103 KB.
// =====================================================================
__global__ void __launch_bounds__(128)
k_op(const _Float16* __restrict__ leftT,   // [R][CO][S]
     const _Float16* __restrict__ rightT,  // [R][CO][S]
     const _Float16* __restrict__ woT,     // [OUT][1024]
     const float* __restrict__ mask,       // [S][R]
     const float* __restrict__ b_o,
     float* __restrict__ out) {            // [R][R][OUT]
    extern __shared__ _Float16 smem[];
    _Float16* pan   = smem;                // [256][PANW]
    _Float16* opbuf = smem + OP_OFF;       // [16][OPW]
    float*    scale = (float*)(smem + SCL_OFF);

    const int i0 = blockIdx.x * 4;
    const int j0 = blockIdx.y * 4;
    const int tid = threadIdx.x;
    const int wv   = tid >> 5;
    const int lane = tid & 31;
    const int lo16 = lane & 15;
    const int hi   = lane >> 4;

    // ---- mask normalizer for the 16 pairs ----
    if (tid < 16) {
        int i = i0 + (tid >> 2), j = j0 + (tid & 3);
        float n = 0.f;
        for (int s = 0; s < S_; ++s)
            n += mask[(size_t)s * R_ + i] * mask[(size_t)s * R_ + j];
        scale[tid] = 1.f / (EPS_ + n);
    }

    // ---- cooperative staging: 64KB (left rows 0..127, right rows 128..255) ----
    {
        const _Float16* gl = leftT  + (size_t)i0 * CO_ * S_;  // 128 rows x 128
        const _Float16* gr = rightT + (size_t)j0 * CO_ * S_;
        #pragma unroll
        for (int it = 0; it < 32; ++it) {          // 4096 chunks of 16B / 128 thr
            int idx   = it * 128 + tid;
            int panel = idx >> 11;                 // 0 = left, 1 = right
            int loc   = idx & 2047;
            int row   = loc >> 4;                  // 0..127
            int col8  = loc & 15;                  // 16B chunk in row
            const _Float16* g = (panel ? gr : gl) + row * S_ + col8 * 8;
            *(v8h*)(pan + (panel * 128 + row) * PANW + col8 * 8) = ldg_h8(g);
        }
    }
    __syncthreads();

    // ---------------- GEMM1: K = S = 128, operands from LDS ----------------
    v8f acc[4][2][2] = {};                 // [jp][mt][nt]
    #pragma unroll
    for (int kc = 0; kc < 4; ++kc) {       // K steps of 32
        v16h af[2];
        #pragma unroll
        for (int mt = 0; mt < 2; ++mt) {   // c-tile (rows wv*32 + mt*16 + lo16)
            const _Float16* pa =
                pan + (wv * 32 + mt * 16 + lo16) * PANW + kc * 32 + hi * 8;
            af[mt] = mk16(*(const v8h*)pa, *(const v8h*)(pa + 16));
        }
        #pragma unroll
        for (int jp = 0; jp < 4; ++jp) {
            v16h bf[2];
            #pragma unroll
            for (int nt = 0; nt < 2; ++nt) { // d-tile
                const _Float16* pb =
                    pan + (128 + jp * 32 + nt * 16 + lo16) * PANW + kc * 32 + hi * 8;
                bf[nt] = mk16(*(const v8h*)pb, *(const v8h*)(pb + 16));
            }
            #pragma unroll
            for (int mt = 0; mt < 2; ++mt)
                #pragma unroll
                for (int nt = 0; nt < 2; ++nt)
                    acc[jp][mt][nt] = __builtin_amdgcn_wmma_f32_16x16x32_f16(
                        false, af[mt], false, bf[nt], (short)0,
                        acc[jp][mt][nt], false, false);
        }
    }

    // spill op (f16) to LDS in GEMM2 A-layout: row = pair p, col = c*32+d
    #pragma unroll
    for (int jp = 0; jp < 4; ++jp) {
        const int p = wv * 4 + jp;
        #pragma unroll
        for (int mt = 0; mt < 2; ++mt)
            #pragma unroll
            for (int nt = 0; nt < 2; ++nt)
                #pragma unroll
                for (int v = 0; v < 8; ++v) {
                    int c = mt * 16 + v + 8 * hi;
                    int d = nt * 16 + lo16;
                    opbuf[p * OPW + c * 32 + d] = (_Float16)acc[jp][mt][nt][v];
                }
    }
    __syncthreads();

    // ---------------- GEMM2: K = 1024 ----------------
    v8f acc2[2] = {};                      // N-tiles wv and wv+4
    for (int k2 = 0; k2 < 32; ++k2) {
        const _Float16* pa = &opbuf[lo16 * OPW + k2 * 32 + hi * 8];
        v16h a = mk16(*(const v8h*)pa, *(const v8h*)(pa + 16));
        #pragma unroll
        for (int t = 0; t < 2; ++t) {
            const int o = (wv + t * 4) * 16 + lo16;
            const _Float16* pb = woT + (size_t)o * 1024 + k2 * 32 + hi * 8;
            if (k2 + 1 < 32) __builtin_prefetch(pb + 32, 0, 1);
            v16h b = mk16(ldg_h8(pb), ldg_h8(pb + 16));
            acc2[t] = __builtin_amdgcn_wmma_f32_16x16x32_f16(
                false, a, false, b, (short)0, acc2[t], false, false);
        }
    }

    // epilogue: + b_o, scale by 1/(eps+norm), store fp32 (coalesced per 16 lanes)
    #pragma unroll
    for (int t = 0; t < 2; ++t) {
        const int o  = (wv + t * 4) * 16 + lo16;
        const float bo = b_o[o];
        #pragma unroll
        for (int v = 0; v < 8; ++v) {
            int p = v + 8 * hi;
            int i_ = i0 + (p >> 2), j_ = j0 + (p & 3);
            out[((size_t)i_ * R_ + j_) * OUT_ + o] = (acc2[t][v] + bo) * scale[p];
        }
    }
}

// =====================================================================
extern "C" void kernel_launch(void* const* d_in, const int* in_sizes, int n_in,
                              void* d_out, int out_size, void* d_ws, size_t ws_size,
                              hipStream_t stream) {
    const float* act     = (const float*)d_in[0];
    const float* mask    = (const float*)d_in[1];
    const float* gamma   = (const float*)d_in[2];
    const float* beta    = (const float*)d_in[3];
    const float* w_left  = (const float*)d_in[4];
    const float* b_left  = (const float*)d_in[5];
    const float* w_right = (const float*)d_in[6];
    const float* b_right = (const float*)d_in[7];
    const float* w_o     = (const float*)d_in[8];
    const float* b_o     = (const float*)d_in[9];
    float* out = (float*)d_out;

    // workspace layout (f16):
    _Float16* ws     = (_Float16*)d_ws;
    _Float16* leftT  = ws;                                   // R*CO*S
    _Float16* rightT = leftT  + (size_t)R_ * CO_ * S_;       // R*CO*S
    _Float16* wlr    = rightT + (size_t)R_ * CO_ * S_;       // 64*C
    _Float16* woT    = wlr    + (size_t)64 * C_;             // OUT*1024

    k_prep<<<576, 256, 0, stream>>>(w_left, w_right, w_o, wlr, woT);
    dim3 g1(S_ / 16, R_);
    k_ln_proj<<<g1, 128, 0, stream>>>(act, mask, gamma, beta, wlr,
                                      b_left, b_right, leftT, rightT);
    dim3 g2(R_ / 4, R_ / 4);
    k_op<<<g2, 128, SMEM_BYTES, stream>>>(leftT, rightT, woT, mask, b_o, out);
}
